// RetLayer_25503515804195
// MI455X (gfx1250) — matrix operand order repaired and checked by reading, not compiled, not run on hardware
//
#include <hip/hip_runtime.h>
#include <hip/hip_bf16.h>

// ---------------------------------------------------------------------------
// MI455X (gfx1250, wave32) implementation.
// scores=Q@K^T (35 GF) and out=scores@V (35 GF) via v_wmma_f32_16x16x32_f16,
// f32 accumulation. Compute bound (AI ~ 670 FLOP/byte). Spill-free register
// budget (waves_per_eu(1)), 4-deep load batches behind sched fences for the
// serial reduction chains, dual accumulator chains for WMMA ILP, transposed
// mask for vector (b128) mask loads, split score tiles across 2 waves with a
// double-buffered LDS relayout, table-driven XPOS, immediate-offset stores.
// ---------------------------------------------------------------------------

#define B_   16
#define TXT_ 32
#define VID_ 2048
#define S_   2080
#define D_   256

typedef __attribute__((ext_vector_type(16))) _Float16 v16h;
typedef __attribute__((ext_vector_type(8)))  _Float16 v8h;
typedef __attribute__((ext_vector_type(8)))  float    v8f;

#if __has_builtin(__builtin_amdgcn_sched_barrier)
#define SCHED_FENCE() __builtin_amdgcn_sched_barrier(0)
#else
#define SCHED_FENCE()
#endif

// workspace layout (bytes); total ~68.3 MB
#define QB_OFF   ((size_t)0)
#define KB_OFF   ((size_t)B_*S_*D_*2)                    // 17,039,360 each
#define VT_OFF   ((size_t)2*B_*S_*D_*2)
#define WT_OFF   ((size_t)3*B_*S_*D_*2)
#define MASK_OFF (WT_OFF + (size_t)3*D_*D_*2)            // transposed mask
#define XPOS_OFF (MASK_OFF + (size_t)VID_*VID_*2)        // 4 planes f32
#define PART_OFF (XPOS_OFF + (size_t)4*VID_*D_*4)

// log2(10000)/128
#define L2T_OVER_HALF 0.103810253f

__device__ __forceinline__ v8f wmma_f16(v16h a, v16h b, v8f c) {
    return __builtin_amdgcn_wmma_f32_16x16x32_f16(
        /*neg_a=*/false, a, /*neg_b=*/false, b,
        /*c_mod=*/(short)0, c, /*reuse_a=*/false, /*reuse_b=*/false);
}

// A-matrix 16x32 f16 layout (ISA 7.12.2): lane row = lane%16; halves e<8 at
// col kk+8*sel, halves e>=8 at col kk+8*sel+16 (sel = lane>=16).
__device__ __forceinline__ v16h loadA_f16(const _Float16* row, int kk, int sel) {
    const _Float16* p = row + kk + 8 * sel;
    v8h lo = *(const v8h*)p;
    v8h hi = *(const v8h*)(p + 16);
    v16h o;
#pragma unroll
    for (int i = 0; i < 8; ++i) { o[i] = lo[i]; o[i + 8] = hi[i]; }
    return o;
}

__device__ __forceinline__ v16h loadA_f32cvt(const float* row, int kk, int sel) {
    const float* p = row + kk + 8 * sel;
    v16h o;
#pragma unroll
    for (int i = 0; i < 8; ++i) { o[i] = (_Float16)p[i]; o[i + 8] = (_Float16)p[16 + i]; }
    return o;
}

// B-matrix 32x16 f16 layout: lane col = lane%16; halves e<8 at K = e + 8*sel,
// halves e>=8 at K = e - 8 + 8*sel + 16 (same K order as the A-matrix).
__device__ __forceinline__ v16h loadB_f16(const _Float16* row, int kk, int sel) {
    const _Float16* p = row + kk + 8 * sel;
    v8h lo = *(const v8h*)p;
    v8h hi = *(const v8h*)(p + 16);
    v16h o;
#pragma unroll
    for (int i = 0; i < 8; ++i) { o[i] = lo[i]; o[i + 8] = hi[i]; }
    return o;
}

// ---------------------------------------------------------------------------
// 1) Video weights: transpose + convert to f16. Wt[p][n][k] = W_p[k][n].
// ---------------------------------------------------------------------------
__global__ void k_wprep(const float* __restrict__ WQ, const float* __restrict__ WK,
                        const float* __restrict__ WV, _Float16* __restrict__ Wt) {
    int tid = blockIdx.x * 256 + threadIdx.x;
    if (tid >= 3 * D_ * D_) return;
    int p = tid >> 16, rem = tid & 65535;
    int n = rem >> 8, k = rem & 255;
    const float* W = (p == 0) ? WQ : ((p == 1) ? WK : WV);
    Wt[tid] = (_Float16)W[k * D_ + n];
}

// ---------------------------------------------------------------------------
// 2) XPOS tables: csq/snq (scale) and csk/snk (1/scale), [VID_][D_] f32 each.
// ---------------------------------------------------------------------------
__global__ void k_xpos(float* __restrict__ tab) {
    int tid = blockIdx.x * 256 + threadIdx.x;   // VID_*D_ threads
    int l = tid >> 8, c = tid & 255;
    int i = c >> 1;
    float sv    = (2.0f * (float)i + 0.4f * (float)D_) / (1.4f * (float)D_);
    float scale = exp2f(((float)l * (1.0f / 512.0f)) * log2f(sv));
    float ang   = (float)l * exp2f(-(float)i * L2T_OVER_HALF);
    float sn = sinf(ang), cs = cosf(ang);
    float inv = 1.0f / scale;
    int n = VID_ * D_;
    tab[tid]         = cs * scale;   // csq
    tab[tid + n]     = sn * scale;   // snq
    tab[tid + 2 * n] = cs * inv;     // csk
    tab[tid + 3 * n] = sn * inv;     // snk
}

// ---------------------------------------------------------------------------
// 3a) Transposed sigmoid mask: maskT[c][r] = sigmoid(learn[r][c]) as f16.
//     Tiled LDS transpose: coalesced loads AND stores. The attention kernel
//     needs 8 consecutive ROWS at a fixed column -> one b128 load per tile.
// ---------------------------------------------------------------------------
__global__ void k_maskT(const float* __restrict__ learn, _Float16* __restrict__ maskT) {
    __shared__ _Float16 tile[32][33];   // +1 pad: no bank conflicts
    int c0 = blockIdx.x * 32, r0 = blockIdx.y * 32;
    int tx = threadIdx.x;               // 0..31
    int ty = threadIdx.y;               // 0..7
#pragma unroll
    for (int i = ty; i < 32; i += 8) {
        float x = learn[(size_t)(r0 + i) * VID_ + c0 + tx];
        tile[i][tx] = (_Float16)(1.0f / (1.0f + __expf(-x)));
    }
    __syncthreads();
#pragma unroll
    for (int i = ty; i < 32; i += 8) {
        maskT[(size_t)(c0 + i) * VID_ + r0 + tx] = tile[tx][i];
    }
}

// ---------------------------------------------------------------------------
// 3b) Sparsity loss: deterministic two-stage reduction of off-diagonal
//     sigmoid (== |(1-I)*sigmoid|).
// ---------------------------------------------------------------------------
__global__ void k_loss_part(const float* __restrict__ learn, float* __restrict__ partial) {
    __shared__ float red[256];
    int t = threadIdx.x;
    float sum = 0.f;
    for (int idx = blockIdx.x * 256 + t; idx < VID_ * VID_; idx += gridDim.x * 256) {
        float sg = 1.0f / (1.0f + __expf(-learn[idx]));
        int r = idx >> 11, c = idx & (VID_ - 1);
        sum += (r == c) ? 0.f : sg;
    }
    red[t] = sum;
    __syncthreads();
    for (int s = 128; s > 0; s >>= 1) {
        if (t < s) red[t] += red[t + s];
        __syncthreads();
    }
    if (t == 0) partial[blockIdx.x] = red[0];
}

__global__ void k_loss_final(const float* __restrict__ partial, float* __restrict__ out_loss) {
    __shared__ float red[256];
    int t = threadIdx.x;
    red[t] = partial[t] + partial[t + 256] + partial[t + 512] + partial[t + 768];
    __syncthreads();
    for (int s = 128; s > 0; s >>= 1) {
        if (t < s) red[t] += red[t + s];
        __syncthreads();
    }
    if (t == 0) out_loss[0] = red[0] * (1.0f / ((float)VID_ * (float)VID_));
}

// ---------------------------------------------------------------------------
// 4) Text projections (tiny: 512 rows). f32 math incl. DETR sine pos embed.
// ---------------------------------------------------------------------------
__global__ void k_txt(const float* __restrict__ src, const float* __restrict__ WQ,
                      const float* __restrict__ WK, const float* __restrict__ WV,
                      _Float16* __restrict__ Qb, _Float16* __restrict__ Kb,
                      _Float16* __restrict__ Vt) {
    __shared__ float s_in[256];   // pos + src
    __shared__ float s_src[256];
    int b = blockIdx.x >> 5;
    int s = blockIdx.x & 31;
    int c = threadIdx.x;
    const float* srow = src + ((size_t)b * S_ + s) * D_;
    float xe  = (float)(s + 1) / (32.0f + 1e-6f) * 6.28318530717958647f;
    float dim = exp2f((float)(c >> 1) * L2T_OVER_HALF);
    float raw = xe / dim;
    float pos = (c & 1) ? cosf(raw) : sinf(raw);
    float sv = srow[c];
    s_src[c] = sv;
    s_in[c]  = sv + pos;
    __syncthreads();
    float q = 0.f, k = 0.f, v = 0.f;
    for (int i = 0; i < D_; ++i) {
        float ti = s_in[i], si = s_src[i];
        q += ti * WQ[i * D_ + c];
        k += ti * WK[i * D_ + c];
        v += si * WV[i * D_ + c];
    }
    size_t ro = ((size_t)b * S_ + s) * D_ + c;
    Qb[ro] = (_Float16)q;
    Kb[ro] = (_Float16)k;
    Vt[((size_t)b * D_ + c) * S_ + s] = (_Float16)v;   // V stored transposed
}

// ---------------------------------------------------------------------------
// 5) Video projections via WMMA. One wave per 16-row tile (2048 tiles).
//    A-frags loaded once, reused for Q/K/V. B-frags batched 4-deep behind
//    sched fences; dual accumulator chains; spill-free register budget.
// ---------------------------------------------------------------------------
__global__ void __attribute__((amdgpu_waves_per_eu(1)))
k_vid(const float* __restrict__ src, const _Float16* __restrict__ Wt,
      const float* __restrict__ xtab,
      _Float16* __restrict__ Qb, _Float16* __restrict__ Kb,
      _Float16* __restrict__ Vt) {
    int lane = threadIdx.x;              // blockDim.x == 32
    int tile = blockIdx.x;               // 0..2047
    int b  = tile >> 7;
    int lt = (tile & 127) << 4;          // local video row base
    int sel = lane >> 4;
    int lr  = lane & 15;

    const float* arow = src + ((size_t)b * S_ + TXT_ + lt + lr) * D_;
    v16h A[8];
#pragma unroll
    for (int kk = 0; kk < 8; ++kk) A[kk] = loadA_f32cvt(arow, kk * 32, sel);

    for (int p = 0; p < 3; ++p) {
        const _Float16* W = Wt + (size_t)p * D_ * D_;
        for (int nt = 0; nt < 16; ++nt) {
            int c = nt * 16 + lr;                        // output column
            const _Float16* wrow = W + (size_t)c * D_;   // Wt is [n][k]
            v8f a0 = {}, a1 = {};
#pragma unroll
            for (int half = 0; half < 2; ++half) {       // 4-deep batches
                v16h Bf[4];
#pragma unroll
                for (int q = 0; q < 4; ++q) Bf[q] = loadB_f16(wrow, (half * 4 + q) * 32, sel);
                SCHED_FENCE();
                a0 = wmma_f16(A[half * 4 + 0], Bf[0], a0);
                a1 = wmma_f16(A[half * 4 + 1], Bf[1], a1);
                a0 = wmma_f16(A[half * 4 + 2], Bf[2], a0);
                a1 = wmma_f16(A[half * 4 + 3], Bf[3], a1);
            }
            v8f acc = a0 + a1;

            if (p < 2) {
                // one base pointer each; per-r accesses become imm offsets
                const float* xb = xtab + (size_t)(2 * p) * VID_ * D_
                                       + (size_t)(lt + 8 * sel) * D_ + c;
                _Float16* db = ((p == 0) ? Qb : Kb)
                             + ((size_t)b * S_ + TXT_ + lt + 8 * sel) * D_ + c;
#pragma unroll
                for (int r = 0; r < 8; ++r) {
                    float cs = xb[r * D_];                 // offset r*1024 B
                    float sn = xb[r * D_ + VID_ * D_];     // +2,097,152 B (imm24)
                    float x  = acc[r];
                    float partner = __shfl_xor(x, 1, 32);  // x[c^1]
                    float res = (c & 1) ? (x * cs + partner * sn)
                                        : (x * cs - partner * sn);
                    db[r * D_] = (_Float16)res;
                }
            } else {
                // transposed V: the 8 values are consecutive -> one b128 store
                v8h pk;
#pragma unroll
                for (int r = 0; r < 8; ++r) pk[r] = (_Float16)acc[r];
                *(v8h*)(Vt + ((size_t)b * D_ + c) * S_ + TXT_ + lt + 8 * sel) = pk;
            }
        }
    }
}

// ---------------------------------------------------------------------------
// 6) Attention: out[b, row0:row0+16, :] = (Q@K^T * mask) @ V.
//    Block = 2 waves; wave w owns output cols [128w, 128w+128) AND computes
//    only score tile t=w. Masked scores go through a double-buffered LDS
//    tile (D-layout -> A-layout relayout), one __syncthreads per chunk.
//    Score loads batched 4-deep behind fences (serial chains); scores@V
//    chains are independent so natural scheduling already overlaps them.
// ---------------------------------------------------------------------------
__global__ void __launch_bounds__(64) __attribute__((amdgpu_waves_per_eu(1)))
k_attn(const _Float16* __restrict__ Qb, const _Float16* __restrict__ Kb,
       const _Float16* __restrict__ Vt, const _Float16* __restrict__ maskT,
       float* __restrict__ out) {
    __shared__ _Float16 sS[2][16 * 32];   // ping-pong masked-score chunk (16x32)
    int w    = threadIdx.x >> 5;
    int lane = threadIdx.x & 31;
    int sel  = lane >> 4;
    int lr   = lane & 15;
    int it   = blockIdx.x;                // 0..129
    int b    = blockIdx.y;                // 0..15
    int row0 = it * 16;
    int colb0 = w * 128;
    bool txtrows = (row0 < TXT_);         // uniform over the block

    const _Float16* qrow = Qb + ((size_t)b * S_ + row0 + lr) * D_;
    v16h Qf[8];
#pragma unroll
    for (int kk = 0; kk < 8; ++kk) Qf[kk] = loadA_f16(qrow, kk * 32, sel);

    v8f zero = {};
    v8f acc[8];
#pragma unroll
    for (int nt = 0; nt < 8; ++nt) acc[nt] = zero;

    int buf = 0;
    for (int j = 0; j < S_; j += 32) {
        if (j + 32 < S_)  // hint next K tile toward the caches
            __builtin_prefetch(Kb + ((size_t)b * S_ + j + 32 + 16 * w + lr) * D_, 0, 1);

        // ---- this wave's score tile (cols j+16w .. j+16w+15), K-dim 256 ----
        const _Float16* krow = Kb + ((size_t)b * S_ + j + 16 * w + lr) * D_;
        v8f s0 = zero, s1 = zero;
#pragma unroll
        for (int half = 0; half < 2; ++half) {
            v16h Kf[4];
#pragma unroll
            for (int q = 0; q < 4; ++q) Kf[q] = loadB_f16(krow, (half * 4 + q) * 32, sel);
            SCHED_FENCE();
            s0 = wmma_f16(Qf[half * 4 + 0], Kf[0], s0);
            s1 = wmma_f16(Qf[half * 4 + 1], Kf[1], s1);
            s0 = wmma_f16(Qf[half * 4 + 2], Kf[2], s0);
            s1 = wmma_f16(Qf[half * 4 + 3], Kf[3], s1);
        }
        v8f sacc = s0 + s1;

        // ---- mask + store this wave's 16x16 half of the chunk ----
        {
            int colabs = j + 16 * w + lr;
            int cc = colabs - TXT_; cc = cc < 0 ? 0 : cc;
            v8h mk = {};
            if (!txtrows)   // 8 consecutive rows, fixed col -> one b128 load
                mk = *(const v8h*)(maskT + (size_t)cc * VID_ + (row0 - TXT_ + 8 * sel));
#pragma unroll
            for (int r = 0; r < 8; ++r) {
                int m = r + 8 * sel;
                float mask = (txtrows || colabs < TXT_) ? 1.0f : (float)mk[r];
                sS[buf][m * 32 + 16 * w + lr] = (_Float16)(sacc[r] * mask);
            }
        }
        __syncthreads();   // both halves of the 16x32 chunk visible

        // ---- re-read masked scores as 16x32 A-matrix fragment ----
        v16h aS;
        {
            const _Float16* pl = &sS[buf][lr * 32 + 8 * sel];
            v8h lo = *(const v8h*)pl;
            v8h hi = *(const v8h*)(pl + 16);
#pragma unroll
            for (int i = 0; i < 8; ++i) { aS[i] = lo[i]; aS[i + 8] = hi[i]; }
        }

        // ---- out += scores(16x32) @ V(32 x this wave's 128 cols) ----
        // 8 independent chains: each WMMA waits only on its own load.
#pragma unroll
        for (int nt = 0; nt < 8; ++nt) {
            int c = colb0 + nt * 16 + lr;
            v16h Vf = loadB_f16(Vt + ((size_t)b * D_ + c) * S_ + j, 0, sel);
            acc[nt] = wmma_f16(aS, Vf, acc[nt]);
        }
        buf ^= 1;   // ping-pong removes the WAR barrier
    }

    // ---- writeback: one base pointer, imm offsets (r*1024 + nt*64 bytes) ----
    float* ob = out + ((size_t)b * S_ + row0 + 8 * sel) * D_ + colb0 + lr;
#pragma unroll
    for (int nt = 0; nt < 8; ++nt) {
#pragma unroll
        for (int r = 0; r < 8; ++r) {
            ob[r * D_ + nt * 16] = acc[nt][r];
        }
    }
}

// ---------------------------------------------------------------------------
extern "C" void kernel_launch(void* const* d_in, const int* in_sizes, int n_in,
                              void* d_out, int out_size, void* d_ws, size_t ws_size,
                              hipStream_t stream) {
    const float* src   = (const float*)d_in[0];
    const float* learn = (const float*)d_in[1];
    const float* WtQ   = (const float*)d_in[2];
    const float* WtK   = (const float*)d_in[3];
    const float* WtV   = (const float*)d_in[4];
    const float* WvQ   = (const float*)d_in[5];
    const float* WvK   = (const float*)d_in[6];
    const float* WvV   = (const float*)d_in[7];
    (void)in_sizes; (void)n_in; (void)ws_size;

    float* out = (float*)d_out;
    char*  ws  = (char*)d_ws;
    _Float16* Qb    = (_Float16*)(ws + QB_OFF);
    _Float16* Kb    = (_Float16*)(ws + KB_OFF);
    _Float16* Vt    = (_Float16*)(ws + VT_OFF);
    _Float16* Wt    = (_Float16*)(ws + WT_OFF);
    _Float16* maskT = (_Float16*)(ws + MASK_OFF);
    float*    xtab  = (float*)(ws + XPOS_OFF);
    float*    partial = (float*)(ws + PART_OFF);
    float*    loss    = out + (out_size - 1);

    k_wprep<<<(3 * D_ * D_ + 255) / 256, 256, 0, stream>>>(WvQ, WvK, WvV, Wt);
    k_xpos<<<VID_ * D_ / 256, 256, 0, stream>>>(xtab);
    dim3 gm(VID_ / 32, VID_ / 32);
    k_maskT<<<gm, dim3(32, 8), 0, stream>>>(learn, maskT);
    k_loss_part<<<1024, 256, 0, stream>>>(learn, partial);
    k_loss_final<<<1, 256, 0, stream>>>(partial, loss);
    k_txt<<<B_ * TXT_, 256, 0, stream>>>(src, WtQ, WtK, WtV, Qb, Kb, Vt);
    k_vid<<<B_ * (VID_ / 16), 32, 0, stream>>>(src, Wt, xtab, Qb, Kb, Vt);
    dim3 ga(S_ / 16, B_);
    k_attn<<<ga, 64, 0, stream>>>(Qb, Kb, Vt, maskT, out);
}
